// OnlineLoss_41300405518929
// MI455X (gfx1250) — matrix operand order repaired and checked by reading, not compiled, not run on hardware
//
#include <hip/hip_runtime.h>
#include <math.h>

#define NN 1024
#define TPB 256

typedef __attribute__((ext_vector_type(16))) __bf16 v16bf;
typedef __attribute__((ext_vector_type(8)))  __bf16 v8bf;
typedef __attribute__((ext_vector_type(8)))  float  v8f;

__device__ __forceinline__ unsigned short f2bf(float f) {
  unsigned int u = __float_as_uint(f);
  u += 0x7FFFu + ((u >> 16) & 1u);          // round to nearest even
  return (unsigned short)(u >> 16);
}

// ---------------------------------------------------------------------------
// WMMA GEMM: C[1024x1024] = epi( A[1024xK] * B ), f32 in/out, bf16 compute.
// BT==0: B is row-major [K][1024]          (C = A @ B)
// BT==1: B is row-major [1024][K]          (C = A @ B^T)
// EPI==0: C = acc ; EPI==1: C = 1 - acc
// Block tile 128x128, K-step 32, 8 waves; each wave owns a 32x64 sub-tile
// (2x4 tiles of 16x16), accumulators in v8f, operands staged in LDS as bf16.
// ---------------------------------------------------------------------------
template<int BT, int EPI>
__global__ __launch_bounds__(256)
void gemm_wmma_bf16(const float* __restrict__ A, const float* __restrict__ B,
                    float* __restrict__ C, int K, int lda, int ldb, int ldc)
{
  constexpr int BM = 128, BN = 128, BK = 32, SB = 48;   // SB: Bt LDS stride (halves)
  __shared__ __align__(16) unsigned short sA[BM * BK];  // A tile, row-major [m][k]
  __shared__ __align__(16) unsigned short sB[BN * SB];  // B tile transposed [n][k]

  const int tid   = threadIdx.x;
  const int tileM = blockIdx.y * BM;
  const int tileN = blockIdx.x * BN;
  const int wave  = tid >> 5;
  const int lane  = tid & 31;
  const int wr    = wave & 3;    // wave row group  -> rows wr*32
  const int wc    = wave >> 2;   // wave col group  -> cols wc*64
  const int h     = lane >> 4;   // lane half
  const int ln    = lane & 15;

  v8f acc[2][4];
#pragma unroll
  for (int mt = 0; mt < 2; ++mt)
#pragma unroll
    for (int nt = 0; nt < 4; ++nt)
#pragma unroll
      for (int r = 0; r < 8; ++r) acc[mt][nt][r] = 0.0f;

  for (int k0 = 0; k0 < K; k0 += BK) {
    // --- stage A tile: 128x32 f32 -> bf16 LDS (16 elems / thread) ---
    {
      const int row = tid >> 1;
      const int seg = (tid & 1) * 16;
      const float* src = A + (size_t)(tileM + row) * lda + k0 + seg;
      unsigned short* dst = sA + row * BK + seg;
#pragma unroll
      for (int e = 0; e < 16; ++e) dst[e] = f2bf(src[e]);
    }
    // --- stage B tile into transposed LDS layout sB[n][k] ---
    if (BT) {
      const int n   = tid >> 1;
      const int seg = (tid & 1) * 16;
      const float* src = B + (size_t)(tileN + n) * ldb + k0 + seg;
      unsigned short* dst = sB + n * SB + seg;
#pragma unroll
      for (int e = 0; e < 16; ++e) dst[e] = f2bf(src[e]);
    } else {
#pragma unroll
      for (int it = 0; it < 16; ++it) {
        int idx = tid + it * 256;
        int kk  = idx >> 7;        // 0..31
        int n   = idx & 127;       // 0..127
        sB[n * SB + kk] = f2bf(B[(size_t)(k0 + kk) * ldb + tileN + n]);
      }
    }
    __syncthreads();

    // --- A fragments: ISA 16-bit A layout: lanes 0-15 K=0..7/16..23, lanes 16-31 +8 ---
    v16bf afr[2];
#pragma unroll
    for (int mt = 0; mt < 2; ++mt) {
      const int row = wr * 32 + mt * 16 + ln;
      const v8bf lo = *(const v8bf*)(const void*)(sA + row * BK + 8 * h);
      const v8bf hi = *(const v8bf*)(const void*)(sA + row * BK + 16 + 8 * h);
      afr[mt] = __builtin_shufflevector(lo, hi, 0,1,2,3,4,5,6,7,8,9,10,11,12,13,14,15);
    }
    // --- B fragments + WMMA: lane n holds K=0..7/16..23 (lanes 0-15), +8 (16-31) ---
#pragma unroll
    for (int nt = 0; nt < 4; ++nt) {
      const int col = wc * 64 + nt * 16 + ln;
      const v8bf lo = *(const v8bf*)(const void*)(sB + col * SB + 8 * h);
      const v8bf hi = *(const v8bf*)(const void*)(sB + col * SB + 16 + 8 * h);
      const v16bf bfr = __builtin_shufflevector(lo, hi, 0,1,2,3,4,5,6,7,8,9,10,11,12,13,14,15);
#pragma unroll
      for (int mt = 0; mt < 2; ++mt) {
        acc[mt][nt] = __builtin_amdgcn_wmma_f32_16x16x32_bf16(
            false, afr[mt], false, bfr, (short)0, acc[mt][nt], false, false);
      }
    }
    __syncthreads();
  }

  // --- store: C/D layout: VGPR r -> M = r + 8*h, N = lane%16 ---
#pragma unroll
  for (int mt = 0; mt < 2; ++mt)
#pragma unroll
    for (int nt = 0; nt < 4; ++nt) {
      const int row0 = tileM + wr * 32 + mt * 16 + 8 * h;
      const int col  = tileN + wc * 64 + nt * 16 + ln;
#pragma unroll
      for (int r = 0; r < 8; ++r) {
        float v = acc[mt][nt][r];
        if (EPI == 1) v = 1.0f - v;
        C[(size_t)(row0 + r) * ldc + col] = v;
      }
    }
}

// ---------------------------------------------------------------------------
// Small helper kernels
// ---------------------------------------------------------------------------
__device__ __forceinline__ float blkSum(float v, float* sh) {
  int t = threadIdx.x;
  sh[t] = v; __syncthreads();
  for (int s = 128; s > 0; s >>= 1) { if (t < s) sh[t] += sh[t + s]; __syncthreads(); }
  float r = sh[0]; __syncthreads(); return r;
}
__device__ __forceinline__ float blkMax(float v, float* sh) {
  int t = threadIdx.x;
  sh[t] = v; __syncthreads();
  for (int s = 128; s > 0; s >>= 1) { if (t < s) sh[t] = fmaxf(sh[t], sh[t + s]); __syncthreads(); }
  float r = sh[0]; __syncthreads(); return r;
}
__device__ __forceinline__ float blkMin(float v, float* sh) {
  int t = threadIdx.x;
  sh[t] = v; __syncthreads();
  for (int s = 128; s > 0; s >>= 1) { if (t < s) sh[t] = fminf(sh[t], sh[t + s]); __syncthreads(); }
  float r = sh[0]; __syncthreads(); return r;
}

__global__ __launch_bounds__(256) void fill_k(float* p, float v, int n) {
  int i = blockIdx.x * blockDim.x + threadIdx.x;
  if (i < n) p[i] = v;
}

// out[i] = (1/NN) * sum_j C[i][j]^2
__global__ __launch_bounds__(256) void rowsq_mean_k(const float* __restrict__ C,
                                                    float* __restrict__ out) {
  __shared__ float red[256];
  const float* row = C + (size_t)blockIdx.x * NN;
  float s = 0.0f;
  for (int j = threadIdx.x; j < NN; j += 256) { float v = row[j]; s += v * v; }
  s = blkSum(s, red);
  if (threadIdx.x == 0) out[blockIdx.x] = s * (1.0f / NN);
}

// grad = 2*(c1v[i]+c2v[j]) - 4*T ; per-row min/max partials
__global__ __launch_bounds__(256) void grad_minmax_k(const float* __restrict__ T,
                                                     const float* __restrict__ c1v,
                                                     const float* __restrict__ c2v,
                                                     float* __restrict__ pmin,
                                                     float* __restrict__ pmax) {
  __shared__ float red[256];
  const int i = blockIdx.x;
  const float c1 = c1v[i];
  const float* row = T + (size_t)i * NN;
  float mn = 3.4e38f, mx = -3.4e38f;
  for (int j = threadIdx.x; j < NN; j += 256) {
    float v = 2.0f * (c1 + c2v[j]) - 4.0f * row[j];
    mn = fminf(mn, v); mx = fmaxf(mx, v);
  }
  mn = blkMin(mn, red);
  mx = blkMax(mx, red);
  if (threadIdx.x == 0) { pmin[i] = mn; pmax[i] = mx; }
}

// sc[0]=gmin sc[1]=gmax sc[2]=eps
__global__ __launch_bounds__(256) void minmax_final_k(const float* __restrict__ pmin,
                                                      const float* __restrict__ pmax,
                                                      float* __restrict__ sc) {
  __shared__ float red[256];
  float mn = 3.4e38f, mx = -3.4e38f;
  for (int j = threadIdx.x; j < NN; j += 256) {
    mn = fminf(mn, pmin[j]); mx = fmaxf(mx, pmax[j]);
  }
  mn = blkMin(mn, red);
  mx = blkMax(mx, red);
  if (threadIdx.x == 0) {
    sc[0] = mn; sc[1] = mx;
    sc[2] = 5e-3f * ((mx - mn) + 1e-12f);   // EPS_REL * Mi.max()
  }
}

// Mi = 2*(c1v[i]+c2v[j]) - 4*T - gmin
__global__ __launch_bounds__(256) void mi_k(const float* __restrict__ T,
                                            const float* __restrict__ c1v,
                                            const float* __restrict__ c2v,
                                            const float* __restrict__ sc,
                                            float* __restrict__ Mi) {
  const int i = blockIdx.x;
  const float c1 = c1v[i], gmin = sc[0];
  const float* row = T + (size_t)i * NN;
  float* orow = Mi + (size_t)i * NN;
  for (int j = threadIdx.x; j < NN; j += 256)
    orow[j] = 2.0f * (c1 + c2v[j]) - 4.0f * row[j] - gmin;
}

__global__ void transpose_k(const float* __restrict__ src, float* __restrict__ dst) {
  __shared__ float tile[32][33];
  const int bx = blockIdx.x * 32, by = blockIdx.y * 32;
  const int tx = threadIdx.x, ty = threadIdx.y;   // 32 x 8
#pragma unroll
  for (int r = 0; r < 32; r += 8)
    tile[ty + r][tx] = src[(size_t)(by + ty + r) * NN + bx + tx];
  __syncthreads();
#pragma unroll
  for (int r = 0; r < 32; r += 8)
    dst[(size_t)(bx + ty + r) * NN + by + tx] = tile[tx][ty + r];
}

// vout[i] = eps*(logu - logsumexp_j((vin[j]-M[i][j])/eps))
__global__ __launch_bounds__(256) void sinkhorn_k(const float* __restrict__ M,
                                                  const float* __restrict__ vin,
                                                  float* __restrict__ vout,
                                                  const float* __restrict__ sc,
                                                  float logu) {
  __shared__ float red[256];
  const int i = blockIdx.x, t = threadIdx.x;
  const float eps = sc[2], inv_eps = 1.0f / eps;
  const float* row = M + (size_t)i * NN;
  float x[4];
  float mx = -3.4e38f;
#pragma unroll
  for (int e = 0; e < 4; ++e) {
    int j = t + e * 256;
    x[e] = (vin[j] - row[j]) * inv_eps;
    mx = fmaxf(mx, x[e]);
  }
  mx = blkMax(mx, red);
  float s = 0.0f;
#pragma unroll
  for (int e = 0; e < 4; ++e) s += __expf(x[e] - mx);
  s = blkSum(s, red);
  if (t == 0) vout[i] = eps * (logu - (mx + __logf(s)));
}

// Gc = exp((f[i]+g[j]-Mi)/eps)
__global__ __launch_bounds__(256) void plan_k(const float* __restrict__ Mi,
                                              const float* __restrict__ f,
                                              const float* __restrict__ g,
                                              const float* __restrict__ sc,
                                              float* __restrict__ Gc) {
  const int i = blockIdx.x;
  const float inv_eps = 1.0f / sc[2];
  const float fi = f[i];
  const float* row = Mi + (size_t)i * NN;
  float* orow = Gc + (size_t)i * NN;
  for (int j = threadIdx.x; j < NN; j += 256)
    orow[j] = __expf((fi + g[j] - row[j]) * inv_eps);
}

// partial sums for line search: s1=TGc*Gc s2=TGc*G s3=TG*Gc s4=TG*G
__global__ __launch_bounds__(256) void ls_part_k(const float* __restrict__ TGc,
                                                 const float* __restrict__ TG,
                                                 const float* __restrict__ Gc,
                                                 const float* __restrict__ G,
                                                 float* p1, float* p2, float* p3, float* p4) {
  __shared__ float red[256];
  const size_t base = (size_t)blockIdx.x * NN;
  float s1 = 0, s2 = 0, s3 = 0, s4 = 0;
  for (int j = threadIdx.x; j < NN; j += 256) {
    float tc = TGc[base + j], tg = TG[base + j];
    float gc = Gc[base + j],  gg = G[base + j];
    s1 += tc * gc; s2 += tc * gg; s3 += tg * gc; s4 += tg * gg;
  }
  s1 = blkSum(s1, red); s2 = blkSum(s2, red);
  s3 = blkSum(s3, red); s4 = blkSum(s4, red);
  if (threadIdx.x == 0) { p1[blockIdx.x] = s1; p2[blockIdx.x] = s2;
                          p3[blockIdx.x] = s3; p4[blockIdx.x] = s4; }
}

// sc[7] = alpha
__global__ __launch_bounds__(256) void ls_final_k(const float* p1, const float* p2,
                                                  const float* p3, const float* p4,
                                                  float* sc) {
  __shared__ float red[256];
  float s1 = 0, s2 = 0, s3 = 0, s4 = 0;
  for (int j = threadIdx.x; j < NN; j += 256) {
    s1 += p1[j]; s2 += p2[j]; s3 += p3[j]; s4 += p4[j];
  }
  s1 = blkSum(s1, red); s2 = blkSum(s2, red);
  s3 = blkSum(s3, red); s4 = blkSum(s4, red);
  if (threadIdx.x == 0) {
    float a = -2.0f * (s1 - s2 - s3 + s4);
    float b = -2.0f * (s2 + s3 - 2.0f * s4);
    float alpha;
    if (a > 0.0f) alpha = fminf(1.0f, fmaxf(0.0f, -b / (2.0f * a)));
    else          alpha = (a + b < 0.0f) ? 1.0f : 0.0f;
    sc[7] = alpha;
  }
}

__global__ __launch_bounds__(256) void update_g_k(float* __restrict__ G,
                                                  const float* __restrict__ Gc,
                                                  const float* __restrict__ sc, int n) {
  const float alpha = sc[7];
  int i = blockIdx.x * blockDim.x + threadIdx.x;
  if (i < n) G[i] += alpha * (Gc[i] - G[i]);
}

// partial of loss = sum (c1v[i]+c2v[j]-2*T) * G
__global__ __launch_bounds__(256) void loss_part_k(const float* __restrict__ T,
                                                   const float* __restrict__ G,
                                                   const float* __restrict__ c1v,
                                                   const float* __restrict__ c2v,
                                                   float* __restrict__ part) {
  __shared__ float red[256];
  const int i = blockIdx.x;
  const float c1 = c1v[i];
  const size_t base = (size_t)i * NN;
  float s = 0.0f;
  for (int j = threadIdx.x; j < NN; j += 256)
    s += (c1 + c2v[j] - 2.0f * T[base + j]) * G[base + j];
  s = blkSum(s, red);
  if (threadIdx.x == 0) part[i] = s;
}

__global__ __launch_bounds__(256) void loss_final_k(const float* __restrict__ part,
                                                    float* __restrict__ out) {
  __shared__ float red[256];
  float s = 0.0f;
  for (int j = threadIdx.x; j < NN; j += 256) s += part[j];
  s = blkSum(s, red);
  if (threadIdx.x == 0) out[0] = s;
}

// ---------------------------------------------------------------------------
extern "C" void kernel_launch(void* const* d_in, const int* in_sizes, int n_in,
                              void* d_out, int out_size, void* d_ws, size_t ws_size,
                              hipStream_t stream) {
  const float* p  = (const float*)d_in[0];   // [1024, D]
  const float* C1 = (const float*)d_in[1];   // [1024, 1024] tree metric
  const int D = in_sizes[0] / NN;            // 512

  const size_t S = (size_t)NN * NN;
  float* ws  = (float*)d_ws;
  float* C2  = ws;            // 1 - p p^T
  float* G   = ws + 1 * S;
  float* Gc  = ws + 2 * S;
  float* TG  = ws + 3 * S;    // C1 @ G  @ C2
  float* TGc = ws + 4 * S;    // C1 @ Gc @ C2
  float* Mi  = ws + 5 * S;
  float* MiT = ws + 6 * S;
  float* P   = ws + 7 * S;    // GEMM intermediate
  float* c1v = ws + 8 * S;
  float* c2v = c1v + NN;
  float* fv  = c1v + 2 * NN;
  float* gv  = c1v + 3 * NN;
  float* pmn = c1v + 4 * NN;
  float* pmx = c1v + 5 * NN;
  float* q1  = c1v + 6 * NN;
  float* q2  = c1v + 7 * NN;
  float* q3  = c1v + 8 * NN;
  float* q4  = c1v + 9 * NN;
  float* sc  = c1v + 10 * NN; // scalars: 0 gmin, 1 gmax, 2 eps, 7 alpha

  const dim3 gGemm(NN / 128, NN / 128), bGemm(256);
  const dim3 gRow(NN), bRow(256);
  const dim3 gTr(NN / 32, NN / 32), bTr(32, 8);
  const int  fillBlocksS = (int)((S + 255) / 256);
  const float LOGU = -6.9314718055994531f;   // log(1/1024)

  // --- setup: C2 = 1 - p p^T ; c1v/c2v ; G0 = 1/(N*N) ---
  gemm_wmma_bf16<1, 1><<<gGemm, bGemm, 0, stream>>>(p, p, C2, D, D, D, NN);
  rowsq_mean_k<<<gRow, bRow, 0, stream>>>(C1, c1v);
  rowsq_mean_k<<<gRow, bRow, 0, stream>>>(C2, c2v);
  fill_k<<<fillBlocksS, 256, 0, stream>>>(G, 1.0f / ((float)NN * (float)NN), (int)S);

  for (int it = 0; it < 30; ++it) {
    // TG = C1 @ G @ C2   (C2 symmetric)
    gemm_wmma_bf16<0, 0><<<gGemm, bGemm, 0, stream>>>(C1, G, P, NN, NN, NN, NN);
    gemm_wmma_bf16<0, 0><<<gGemm, bGemm, 0, stream>>>(P, C2, TG, NN, NN, NN, NN);
    // eps, Mi, MiT
    grad_minmax_k<<<gRow, bRow, 0, stream>>>(TG, c1v, c2v, pmn, pmx);
    minmax_final_k<<<1, 256, 0, stream>>>(pmn, pmx, sc);
    mi_k<<<gRow, bRow, 0, stream>>>(TG, c1v, c2v, sc, Mi);
    transpose_k<<<gTr, bTr, 0, stream>>>(Mi, MiT);
    // Sinkhorn LMO
    fill_k<<<(NN + 255) / 256, 256, 0, stream>>>(gv, 0.0f, NN);
    for (int s = 0; s < 100; ++s) {
      sinkhorn_k<<<gRow, bRow, 0, stream>>>(Mi,  gv, fv, sc, LOGU);
      sinkhorn_k<<<gRow, bRow, 0, stream>>>(MiT, fv, gv, sc, LOGU);
    }
    plan_k<<<gRow, bRow, 0, stream>>>(Mi, fv, gv, sc, Gc);
    // line search: TGc = C1 @ Gc @ C2
    gemm_wmma_bf16<0, 0><<<gGemm, bGemm, 0, stream>>>(C1, Gc, P, NN, NN, NN, NN);
    gemm_wmma_bf16<0, 0><<<gGemm, bGemm, 0, stream>>>(P, C2, TGc, NN, NN, NN, NN);
    ls_part_k<<<gRow, bRow, 0, stream>>>(TGc, TG, Gc, G, q1, q2, q3, q4);
    ls_final_k<<<1, 256, 0, stream>>>(q1, q2, q3, q4, sc);
    update_g_k<<<fillBlocksS, 256, 0, stream>>>(G, Gc, sc, (int)S);
  }

  // final loss = sum( (constC - 2*C1 G C2) * G )
  gemm_wmma_bf16<0, 0><<<gGemm, bGemm, 0, stream>>>(C1, G, P, NN, NN, NN, NN);
  gemm_wmma_bf16<0, 0><<<gGemm, bGemm, 0, stream>>>(P, C2, TG, NN, NN, NN, NN);
  loss_part_k<<<gRow, bRow, 0, stream>>>(TG, G, c1v, c2v, q1);
  loss_final_k<<<1, 256, 0, stream>>>(q1, (float*)d_out);
}
